// Pointnet2_Memory_90486370992544
// MI455X (gfx1250) — matrix operand order repaired and checked by reading, not compiled, not run on hardware
//
#include <hip/hip_runtime.h>
#include <hip/hip_bf16.h>
#include <stdint.h>
#include <stddef.h>

// ---------------------------------------------------------------------------
// PointNet++ (B=2, N=16384, 13 classes) forward for gfx1250 (MI455X).
// All MLP GEMMs -> v_wmma_f32_16x16x32_f16 (one wave per 16x16 tile).
// Weights pre-packed on device into WMMA B-fragment order (contiguous 32B/lane).
// Input flattening assumed in setup_inputs() dict order:
//   d_in[0]=pos d_in[1]=color d_in[2]=normal, then params in insertion order
//   sa1..sa4, fp1..fp4, out; each layer contributes (W, b).
// ---------------------------------------------------------------------------

typedef _Float16 h16;
typedef __attribute__((ext_vector_type(16))) _Float16 v16h_t;
typedef __attribute__((ext_vector_type(8)))  _Float16 v8h_t;
typedef __attribute__((ext_vector_type(8)))  float    v8f_t;

#define BATCH 2
#define NPTS  16384
#define NCLS  13
#define KNB   32

static inline int cdiv(int a, int b) { return (a + b - 1) / b; }

// ---------------------------------------------------------------------------
// Build x0 = [color(3), height(1), normal(3)]  (B*N rows, 7 cols, f32)
// ---------------------------------------------------------------------------
__global__ void k_build_x0(const float* __restrict__ color,
                           const float* __restrict__ pos,
                           const float* __restrict__ normal,
                           float* __restrict__ x0, int total) {
  int t = blockIdx.x * blockDim.x + threadIdx.x;
  if (t >= total) return;
  x0[t * 7 + 0] = color[t * 3 + 0];
  x0[t * 7 + 1] = color[t * 3 + 1];
  x0[t * 7 + 2] = color[t * 3 + 2];
  x0[t * 7 + 3] = pos[t * 3 + 2];
  x0[t * 7 + 4] = normal[t * 3 + 0];
  x0[t * 7 + 5] = normal[t * 3 + 1];
  x0[t * 7 + 6] = normal[t * 3 + 2];
}

// ---------------------------------------------------------------------------
// Farthest point sampling: one block per batch. Sequential ns-1 argmax steps,
// per-thread resident distance chunks, block reduction, first-index tie-break.
// Writes sampled center positions directly (fidx kept for completeness).
// ---------------------------------------------------------------------------
__global__ __launch_bounds__(256) void k_fps(const float* __restrict__ pts,
                                             int n, int ns,
                                             int* __restrict__ fidx,
                                             float* __restrict__ cent) {
  const int b = blockIdx.x, tid = threadIdx.x;
  const float* P = pts + (size_t)b * n * 3;
  int* FI = fidx + (size_t)b * ns;
  float* CE = cent + (size_t)b * ns * 3;
  const int ppt = n / 256;  // 64,16,4,1
  float ld[64];
  for (int p = 0; p < ppt; ++p) ld[p] = 1e10f;
  __shared__ float sx, sy, sz;
  __shared__ float sv[256];
  __shared__ int   si[256];
  __shared__ int   slast;
  if (tid == 0) {
    FI[0] = 0; CE[0] = P[0]; CE[1] = P[1]; CE[2] = P[2]; slast = 0;
  }
  __syncthreads();
  for (int s = 1; s < ns; ++s) {
    if (tid == 0) { int l = slast; sx = P[l*3]; sy = P[l*3+1]; sz = P[l*3+2]; }
    __syncthreads();
    float best = -1.0f; int bi = tid * ppt;
    for (int p = 0; p < ppt; ++p) {
      int i = tid * ppt + p;
      float dx = P[i*3] - sx, dy = P[i*3+1] - sy, dz = P[i*3+2] - sz;
      float d = dx*dx + dy*dy + dz*dz;
      float m = fminf(ld[p], d);
      ld[p] = m;
      if (m > best) { best = m; bi = i; }
    }
    sv[tid] = best; si[tid] = bi;
    __syncthreads();
    for (int st = 128; st > 0; st >>= 1) {
      if (tid < st) {
        float ov = sv[tid + st]; int oi = si[tid + st];
        if (ov > sv[tid] || (ov == sv[tid] && oi < si[tid])) { sv[tid] = ov; si[tid] = oi; }
      }
      __syncthreads();
    }
    if (tid == 0) {
      int nx = si[0]; slast = nx; FI[s] = nx;
      CE[s*3] = P[nx*3]; CE[s*3+1] = P[nx*3+1]; CE[s*3+2] = P[nx*3+2];
    }
    __syncthreads();
  }
}

// ---------------------------------------------------------------------------
// Ball query: one wave32 per center. Reproduces sort(key)[:k] semantics:
// first KNB indices (ascending) with d2<=r2, padded with the first hit.
// ---------------------------------------------------------------------------
__global__ __launch_bounds__(256) void k_ballq(const float* __restrict__ pts,
                                               const float* __restrict__ cent,
                                               int n, int ns, float r2,
                                               int* __restrict__ gidx) {
  int gt = blockIdx.x * blockDim.x + threadIdx.x;
  int w = gt >> 5, lane = gt & 31;
  if (w >= BATCH * ns) return;
  int b = w / ns, c = w % ns;
  const float* P = pts + (size_t)b * n * 3;
  const float* CE = cent + ((size_t)b * ns + c) * 3;
  float cx = CE[0], cy = CE[1], cz = CE[2];
  int* G = gidx + ((size_t)b * ns + c) * KNB;
  int cnt = 0;
  for (int base = 0; base < n && cnt < KNB; base += 32) {
    int i = base + lane;
    float dx = P[i*3] - cx, dy = P[i*3+1] - cy, dz = P[i*3+2] - cz;
    bool in = (dx*dx + dy*dy + dz*dz) <= r2;
    unsigned mm = (unsigned)__ballot(in);
    if (lane == 0) {
      while (mm && cnt < KNB) {
        int bit = __ffs((int)mm) - 1;
        G[cnt++] = base + bit;
        mm &= mm - 1;
      }
    }
    cnt = __shfl(cnt, 0, 32);
  }
  if (lane == 0) {
    if (cnt == 0) { G[0] = 0; cnt = 1; }
    int f = G[0];
    for (int j = cnt; j < KNB; ++j) G[j] = f;
  }
}

// ---------------------------------------------------------------------------
// Grouping: build f16 A rows [rel(3)/radius , feats(Cf) , zero-pad -> Kpad]
// ---------------------------------------------------------------------------
__global__ void k_group(const float* __restrict__ pts,
                        const float* __restrict__ feats,
                        const float* __restrict__ cent,
                        const int* __restrict__ gidx,
                        int n, int ns, int Cf, int Kpad, float inv_r,
                        h16* __restrict__ A) {
  int t = blockIdx.x * blockDim.x + threadIdx.x;
  int total = BATCH * ns * KNB;
  if (t >= total) return;
  int c = (t / KNB) % ns;
  int b = t / (KNB * ns);
  int idx = gidx[t];
  const float* P = pts + ((size_t)b * n + idx) * 3;
  const float* F = feats + ((size_t)b * n + idx) * Cf;
  const float* CE = cent + ((size_t)b * ns + c) * 3;
  h16* row = A + (size_t)t * Kpad;
  row[0] = (h16)((P[0] - CE[0]) * inv_r);
  row[1] = (h16)((P[1] - CE[1]) * inv_r);
  row[2] = (h16)((P[2] - CE[2]) * inv_r);
  int k = 3;
  for (int q = 0; q < Cf; ++q) row[k++] = (h16)F[q];
  for (; k < Kpad; ++k) row[k] = (h16)0.0f;
}

// ---------------------------------------------------------------------------
// Pack weights W(O x C) into WMMA B-fragment order:
//   blob = (ks, to): 32 lanes x 16 halfs contiguous;
//   lane -> col o = to*16 + lane%16; element j -> k = ks*32 + (lane>=16?16:0) + j
// Packed size = Kpad * O halfs.
// ---------------------------------------------------------------------------
__global__ void k_pack_w(const float* __restrict__ W, h16* __restrict__ dst,
                         int C, int O, int Kpad) {
  int t = blockIdx.x * blockDim.x + threadIdx.x;
  int total = Kpad * O;
  if (t >= total) return;
  int j = t & 15;
  int lane = (t >> 4) & 31;
  int blob = t >> 9;
  int nko = O >> 4;
  int ks = blob / nko, to = blob % nko;
  int o = to * 16 + (lane & 15);
  int k = ks * 32 + ((lane >> 4) ? 16 : 0) + j;
  float v = (k < C && o < O) ? W[(size_t)o * C + k] : 0.0f;
  dst[t] = (h16)v;
}

// ---------------------------------------------------------------------------
// WMMA GEMM: D[R x O] = act(A[R x Kpad] * Wpacked + bias). One wave per tile.
// A fragment: two contiguous 16B loads per lane per K-step (k = 16*hid + j, same order as packed B).
// B fragment: one contiguous 32B read per lane from packed weights.
// ---------------------------------------------------------------------------
__global__ __launch_bounds__(32) void k_wmma_gemm(
    const h16* __restrict__ A, const h16* __restrict__ Wp,
    const float* __restrict__ bias, void* __restrict__ out,
    int R, int Kpad, int Opad, int relu, int out_f32) {
  const int lane = threadIdx.x & 31;
  const int tm = blockIdx.x;
  const int to = blockIdx.y;
  const int hid = lane >> 4;   // 0: lanes 0-15, 1: lanes 16-31
  const int lm  = lane & 15;
  const int nko = Opad >> 4;
  (void)R;
  v8f_t acc = {};
  const h16* arow = A + (size_t)(tm * 16 + lm) * Kpad;
  const h16* bblk = Wp + (size_t)to * 512 + (size_t)lane * 16;
  const size_t bstep = (size_t)nko * 512;
  for (int ks = 0; ks < Kpad; ks += 32) {
    v8h_t alo = *reinterpret_cast<const v8h_t*>(arow + hid * 16);
    v8h_t ahi = *reinterpret_cast<const v8h_t*>(arow + hid * 16 + 8);
    v8h_t blo = *reinterpret_cast<const v8h_t*>(bblk);
    v8h_t bhi = *reinterpret_cast<const v8h_t*>(bblk + 8);
    v16h_t a, b;
#pragma unroll
    for (int j = 0; j < 8; ++j) {
      a[j] = alo[j]; a[8 + j] = ahi[j];
      b[j] = blo[j]; b[8 + j] = bhi[j];
    }
    acc = __builtin_amdgcn_wmma_f32_16x16x32_f16(
        false, a, false, b, (short)0, acc, false, false);
    arow += 32;
    bblk += bstep;
    __builtin_prefetch(arow, 0, 0);      // global_prefetch_b8
  }
  float bv = bias[to * 16 + lm];
  int row0 = tm * 16 + hid * 8;
  int col = to * 16 + lm;
  if (out_f32) {
    float* O = (float*)out;
#pragma unroll
    for (int r = 0; r < 8; ++r) {
      float v = acc[r] + bv;
      if (relu) v = fmaxf(v, 0.0f);
      O[(size_t)(row0 + r) * Opad + col] = v;
    }
  } else {
    h16* O = (h16*)out;
#pragma unroll
    for (int r = 0; r < 8; ++r) {
      float v = acc[r] + bv;
      if (relu) v = fmaxf(v, 0.0f);
      O[(size_t)(row0 + r) * Opad + col] = (h16)v;
    }
  }
}

// ---------------------------------------------------------------------------
// Max-pool over the KNB group dimension (f16 in, f32 out)
// ---------------------------------------------------------------------------
__global__ void k_maxpool(const h16* __restrict__ H, float* __restrict__ X,
                          int ns, int O) {
  int t = blockIdx.x * blockDim.x + threadIdx.x;
  int total = BATCH * ns * O;
  if (t >= total) return;
  int o = t % O;
  int cc = t / O;  // b*ns + c
  const h16* base = H + ((size_t)cc * KNB) * O + o;
  float m = -1e30f;
  for (int j = 0; j < KNB; ++j) m = fmaxf(m, (float)base[(size_t)j * O]);
  X[t] = m;
}

// ---------------------------------------------------------------------------
// FP: 3-NN inverse-distance interpolation + concat -> f16 A rows
// row = [interp(C2), f1(C1), zero-pad -> Kpad]
// ---------------------------------------------------------------------------
__global__ void k_interp(const float* __restrict__ p1, const float* __restrict__ f1, int C1,
                         const float* __restrict__ p2, const float* __restrict__ f2, int C2,
                         int n1, int n2, int Kpad, h16* __restrict__ A) {
  int t = blockIdx.x * blockDim.x + threadIdx.x;
  int total = BATCH * n1;
  if (t >= total) return;
  int b = t / n1;
  const float* P = p1 + (size_t)t * 3;
  float px = P[0], py = P[1], pz = P[2];
  const float* Q = p2 + (size_t)b * n2 * 3;
  float d0 = 1e30f, d1 = 1e30f, d2v = 1e30f;
  int i0 = 0, i1 = 0, i2 = 0;
  for (int q = 0; q < n2; ++q) {
    float dx = Q[q*3] - px, dy = Q[q*3+1] - py, dz = Q[q*3+2] - pz;
    float d = dx*dx + dy*dy + dz*dz;
    if (d < d0)       { d2v = d1; i2 = i1; d1 = d0; i1 = i0; d0 = d; i0 = q; }
    else if (d < d1)  { d2v = d1; i2 = i1; d1 = d; i1 = q; }
    else if (d < d2v) { d2v = d; i2 = q; }
  }
  float w0 = 1.0f / (d0 + 1e-8f), w1 = 1.0f / (d1 + 1e-8f), w2 = 1.0f / (d2v + 1e-8f);
  float ws = w0 + w1 + w2;
  w0 /= ws; w1 /= ws; w2 /= ws;
  const float* F0 = f2 + ((size_t)b * n2 + i0) * C2;
  const float* F1 = f2 + ((size_t)b * n2 + i1) * C2;
  const float* F2 = f2 + ((size_t)b * n2 + i2) * C2;
  h16* row = A + (size_t)t * Kpad;
  for (int c = 0; c < C2; ++c) row[c] = (h16)(w0 * F0[c] + w1 * F1[c] + w2 * F2[c]);
  const float* Fs = f1 + (size_t)t * C1;
  for (int c = 0; c < C1; ++c) row[C2 + c] = (h16)Fs[c];
  for (int c = C2 + C1; c < Kpad; ++c) row[c] = (h16)0.0f;
}

// ---------------------------------------------------------------------------
// Per-batch channel max over all N points
// ---------------------------------------------------------------------------
__global__ void k_chanmax(const float* __restrict__ X, float* __restrict__ M,
                          int n, int C) {
  int t = blockIdx.x * blockDim.x + threadIdx.x;
  if (t >= BATCH * C) return;
  int b = t / C, c = t % C;
  const float* x = X + (size_t)b * n * C + c;
  float m = -1e30f;
  for (int i = 0; i < n; ++i) m = fmaxf(m, x[(size_t)i * C]);
  M[t] = m;
}

// ---------------------------------------------------------------------------
// Head input: [x(C), broadcast chan-max(C)] -> f16 rows of width 2C
// ---------------------------------------------------------------------------
__global__ void k_head_in(const float* __restrict__ X, const float* __restrict__ M,
                          h16* __restrict__ A, int n, int C) {
  int t = blockIdx.x * blockDim.x + threadIdx.x;
  if (t >= BATCH * n) return;
  int b = t / n;
  const float* x = X + (size_t)t * C;
  const float* m = M + (size_t)b * C;
  h16* row = A + (size_t)t * (2 * C);
  for (int c = 0; c < C; ++c) { row[c] = (h16)x[c]; row[C + c] = (h16)m[c]; }
}

// ---------------------------------------------------------------------------
// Head output: y[b,o,i] = h[b,i,:] . W2[o,:] + b2[o]   (O=13, scalar kernel)
// ---------------------------------------------------------------------------
__global__ void k_head_out(const h16* __restrict__ H, const float* __restrict__ W2,
                           const float* __restrict__ b2, float* __restrict__ out,
                           int n) {
  int t = blockIdx.x * blockDim.x + threadIdx.x;
  int total = BATCH * NCLS * n;
  if (t >= total) return;
  int i = t % n;
  int o = (t / n) % NCLS;
  int b = t / (n * NCLS);
  const h16* h = H + ((size_t)b * n + i) * 256;
  const float* w = W2 + (size_t)o * 256;
  float acc = b2[o];
  for (int c = 0; c < 256; ++c) acc += (float)h[c] * w[c];
  out[t] = acc;
}

// ---------------------------------------------------------------------------
// Host orchestration
// ---------------------------------------------------------------------------
struct LayerCfg { int wi; int C; int O; int Kpad; };
static const LayerCfg LY[21] = {
  {3, 10, 32, 32},   {5, 32, 32, 32},   {7, 32, 64, 32},     // sa1
  {9, 67, 64, 96},   {11, 64, 64, 64},  {13, 64, 128, 64},   // sa2
  {15, 131, 128, 160}, {17, 128, 128, 128}, {19, 128, 256, 128}, // sa3
  {21, 259, 256, 288}, {23, 256, 256, 256}, {25, 256, 512, 256}, // sa4
  {27, 768, 256, 768}, {29, 256, 256, 256},                  // fp1
  {31, 384, 256, 384}, {33, 256, 256, 256},                  // fp2
  {35, 320, 256, 320}, {37, 256, 128, 256},                  // fp3
  {39, 135, 128, 160}, {41, 128, 128, 128},                  // fp4
  {43, 256, 256, 256},                                       // head L1
};

extern "C" void kernel_launch(void* const* d_in, const int* in_sizes, int n_in,
                              void* d_out, int out_size, void* d_ws, size_t ws_size,
                              hipStream_t stream) {
  (void)in_sizes; (void)n_in; (void)out_size; (void)ws_size;
  const float* pos    = (const float*)d_in[0];
  const float* color  = (const float*)d_in[1];
  const float* normal = (const float*)d_in[2];

  char* wsb = (char*)d_ws;
  size_t off = 0;
  auto alloc = [&](size_t bytes) -> char* {
    off = (off + 255) & ~(size_t)255;
    char* p = wsb + off;
    off += bytes;
    return p;
  };

  float* x0   = (float*)alloc((size_t)BATCH * NPTS * 7 * 4);
  float* pos1 = (float*)alloc((size_t)BATCH * 4096 * 3 * 4);
  float* x1   = (float*)alloc((size_t)BATCH * 4096 * 64 * 4);
  float* pos2 = (float*)alloc((size_t)BATCH * 1024 * 3 * 4);
  float* x2   = (float*)alloc((size_t)BATCH * 1024 * 128 * 4);
  float* pos3 = (float*)alloc((size_t)BATCH * 256 * 3 * 4);
  float* x3   = (float*)alloc((size_t)BATCH * 256 * 256 * 4);
  float* pos4 = (float*)alloc((size_t)BATCH * 64 * 3 * 4);
  float* x4   = (float*)alloc((size_t)BATCH * 64 * 512 * 4);
  float* x3n  = (float*)alloc((size_t)BATCH * 256 * 256 * 4);
  float* x2n  = (float*)alloc((size_t)BATCH * 1024 * 256 * 4);
  float* x1n  = (float*)alloc((size_t)BATCH * 4096 * 128 * 4);
  float* x0n  = (float*)alloc((size_t)BATCH * NPTS * 128 * 4);
  float* xmax = (float*)alloc((size_t)BATCH * 128 * 4);
  int*   fidx = (int*)alloc((size_t)BATCH * 4096 * 4);
  int*   gidx = (int*)alloc((size_t)BATCH * 4096 * KNB * 4);
  h16*   Ah   = (h16*)alloc((size_t)8388608 * 2);    // max A: 262144x32 / 32768x256
  h16*   H1h  = (h16*)alloc((size_t)16777216 * 2);   // max hidden: 262144x64
  h16*   H2h  = (h16*)alloc((size_t)8388608 * 2);
  h16*   Wp   = (h16*)alloc((size_t)1005568 * 2);    // sum Kpad*O over all layers

  // --- pack all WMMA weights -----------------------------------------------
  size_t wpoff[21];
  {
    size_t acc = 0;
    for (int i = 0; i < 21; ++i) {
      wpoff[i] = acc;
      acc += (size_t)LY[i].Kpad * LY[i].O;
      int total = LY[i].Kpad * LY[i].O;
      k_pack_w<<<cdiv(total, 256), 256, 0, stream>>>(
          (const float*)d_in[LY[i].wi], Wp + wpoff[i], LY[i].C, LY[i].O, LY[i].Kpad);
    }
  }

  auto gemm = [&](const h16* A, int li, void* out, int R, int f32out) {
    const LayerCfg& L = LY[li];
    dim3 g(R / 16, L.O / 16);
    k_wmma_gemm<<<g, 32, 0, stream>>>(
        A, Wp + wpoff[li], (const float*)d_in[L.wi + 1], out,
        R, L.Kpad, L.O, /*relu=*/1, f32out);
  };

  // --- stage 0: input features ---------------------------------------------
  k_build_x0<<<cdiv(BATCH * NPTS, 256), 256, 0, stream>>>(color, pos, normal, x0,
                                                          BATCH * NPTS);

  // --- SA stages ------------------------------------------------------------
  struct SaCfg {
    int n, ns, Cf, Kpad, li; float r;
    const float* pin; const float* fin; float* pout; float* fout;
  };
  SaCfg sas[4] = {
    {16384, 4096, 7,   32,  0, 0.05f, pos,  x0, pos1, x1},
    {4096,  1024, 64,  96,  3, 0.10f, pos1, x1, pos2, x2},
    {1024,  256,  128, 160, 6, 0.20f, pos2, x2, pos3, x3},
    {256,   64,   256, 288, 9, 0.40f, pos3, x3, pos4, x4},
  };
  for (int s = 0; s < 4; ++s) {
    const SaCfg& c = sas[s];
    k_fps<<<BATCH, 256, 0, stream>>>(c.pin, c.n, c.ns, fidx, c.pout);
    k_ballq<<<(BATCH * c.ns) / 8, 256, 0, stream>>>(c.pin, c.pout, c.n, c.ns,
                                                    c.r * c.r, gidx);
    int Rg = BATCH * c.ns * KNB;
    k_group<<<cdiv(Rg, 256), 256, 0, stream>>>(c.pin, c.fin, c.pout, gidx,
                                               c.n, c.ns, c.Cf, c.Kpad,
                                               1.0f / c.r, Ah);
    gemm(Ah,  c.li + 0, H1h, Rg, 0);
    gemm(H1h, c.li + 1, H2h, Rg, 0);
    gemm(H2h, c.li + 2, H1h, Rg, 0);
    int O2 = LY[c.li + 2].O;
    k_maxpool<<<cdiv(BATCH * c.ns * O2, 256), 256, 0, stream>>>(H1h, c.fout,
                                                                c.ns, O2);
  }

  // --- FP stages ------------------------------------------------------------
  struct FpCfg {
    int n1, n2, C1, C2, Kpad, li;
    const float* p1; const float* f1; const float* p2; const float* f2;
    float* fout;
  };
  FpCfg fps[4] = {
    {256,   64,   256, 512, 768, 12, pos3, x3, pos4, x4,  x3n},
    {1024,  256,  128, 256, 384, 14, pos2, x2, pos3, x3n, x2n},
    {4096,  1024, 64,  256, 320, 16, pos1, x1, pos2, x2n, x1n},
    {16384, 4096, 7,   128, 160, 18, pos,  x0, pos1, x1n, x0n},
  };
  for (int s = 0; s < 4; ++s) {
    const FpCfg& c = fps[s];
    k_interp<<<cdiv(BATCH * c.n1, 256), 256, 0, stream>>>(
        c.p1, c.f1, c.C1, c.p2, c.f2, c.C2, c.n1, c.n2, c.Kpad, Ah);
    int R = BATCH * c.n1;
    gemm(Ah,  c.li + 0, H1h, R, 0);
    gemm(H1h, c.li + 1, c.fout, R, 1);   // f32 out (relu on all mlp layers)
  }

  // --- head -----------------------------------------------------------------
  k_chanmax<<<1, 256, 0, stream>>>(x0n, xmax, NPTS, 128);
  k_head_in<<<cdiv(BATCH * NPTS, 256), 256, 0, stream>>>(x0n, xmax, Ah, NPTS, 128);
  gemm(Ah, 20, H1h, BATCH * NPTS, 0);    // 256 -> 256 relu, f16 out
  k_head_out<<<cdiv(BATCH * NCLS * NPTS, 256), 256, 0, stream>>>(
      H1h, (const float*)d_in[45], (const float*)d_in[46], (float*)d_out, NPTS);
}
